// DLPTLayer_PreLN_36550171688960
// MI455X (gfx1250) — matrix operand order repaired and checked by reading, not compiled, not run on hardware
//
#include <hip/hip_runtime.h>
#include <hip/hip_bf16.h>

// ---------------------------------------------------------------------------
// DLPT PreLN layer for MI455X (gfx1250, wave32, WMMA 16x16x32 f16).
// Every wave owns 32 points: one S=32 cluster, or TWO S=16 clusters
// (lanes 0-15 / 16-31). All de-sized GEMMs are MT=2 WMMA; the S=16 case
// uses a block-diagonal 32x32 attention matrix so attn@V is a plain GEMM.
// Weights are pre-converted once to f16, transposed + K-padded (one aligned
// 32B load per B fragment); all small f32 params are packed into a single
// buffer so the kernel carries only two param pointers (no SGPR spills).
// ---------------------------------------------------------------------------

typedef __attribute__((ext_vector_type(16))) _Float16 v16h;
typedef __attribute__((ext_vector_type(8)))  _Float16 v8h;
typedef __attribute__((ext_vector_type(8)))  float    v8f;

#define DEV static __device__ __forceinline__

DEV v8f zero8() {
  v8f z;
#pragma unroll
  for (int r = 0; r < 8; ++r) z[r] = 0.f;
  return z;
}

DEV v16h cat8(v8h lo, v8h hi) {
  v16h a;
#pragma unroll
  for (int j = 0; j < 8; ++j) { a[j] = lo[j]; a[8 + j] = hi[j]; }
  return a;
}

DEV v8f wmma_f16(v16h a, v16h b, v8f c) {
  return __builtin_amdgcn_wmma_f32_16x16x32_f16(
      /*neg_a=*/false, a, /*neg_b=*/false, b,
      /*c_mod=*/(short)0, c, /*reuse_a=*/false, /*reuse_b=*/false);
}

// A fragment: 16x32 f16 tile from LDS row-major [rows][ld], ld % 16 == 0.
DEV v16h load_a16(const _Float16* sm, int m0, int k0, int ld, int lane) {
  int row = m0 + (lane & 15);
  int hi  = lane >> 4;
  const _Float16* p = sm + row * ld + k0 + hi * 16;
  v8h c0 = *(const v8h*)(p);
  v8h c1 = *(const v8h*)(p + 8);
  return cat8(c0, c1);
}

// B fragment from K-contiguous storage X[n][ld] (B[k][n] = X[n][k]).
DEV v16h load_bt16(const _Float16* sm, int k0, int n0, int ld, int lane) {
  int n  = n0 + (lane & 15);
  int kb = k0 + ((lane >> 4) << 4);
  return *(const v16h*)(sm + n * ld + kb);
}

// B fragment from pre-transposed, pre-padded f16 weight wT[n][kpad] in global.
DEV v16h load_bwT(const _Float16* __restrict__ wt, int k0, int n0, int kpad,
                  int lane) {
  int n  = n0 + (lane & 15);
  int kb = k0 + ((lane >> 4) << 4);
  return *(const v16h*)(wt + n * kpad + kb);
}

// C/D fragment (f32): VGPR r -> row m0 + (lane>=16 ? 8 : 0) + r, col = lane&15.
DEV void store_c16(v8f c, _Float16* sm, int m0, int n0, int ld, int lane,
                   float scale) {
  int col = n0 + (lane & 15);
  int rb  = m0 + ((lane >> 4) << 3);
#pragma unroll
  for (int r = 0; r < 8; ++r) sm[(rb + r) * ld + col] = (_Float16)(c[r] * scale);
}

// Transposed C store: (row,col) -> sm[col*ldT + row]; one aligned 16B store.
DEV void store_c16T(v8f c, _Float16* sm, int m0, int n0, int ldT, int lane,
                    float scale) {
  int col = n0 + (lane & 15);
  int rb  = m0 + ((lane >> 4) << 3);
  v8h val;
#pragma unroll
  for (int r = 0; r < 8; ++r) val[r] = (_Float16)(c[r] * scale);
  *(v8h*)(sm + col * ldT + rb) = val;
}

DEV void store_c32(v8f c, float* sm, int m0, int n0, int ld, int lane) {
  int col = n0 + (lane & 15);
  int rb  = m0 + ((lane >> 4) << 3);
#pragma unroll
  for (int r = 0; r < 8; ++r) sm[(rb + r) * ld + col] = c[r];
}

template <int D>
DEV void ln_relu(float* x, const float* __restrict__ g,
                 const float* __restrict__ e) {
  float mu = 0.f;
#pragma unroll
  for (int j = 0; j < D; ++j) mu += x[j];
  mu *= (1.f / D);
  float var = 0.f;
#pragma unroll
  for (int j = 0; j < D; ++j) { float d = x[j] - mu; var += d * d; }
  var *= (1.f / D);
  float rs = rsqrtf(var + 1e-5f);
#pragma unroll
  for (int j = 0; j < D; ++j) {
    float v = (x[j] - mu) * rs * g[j] + e[j];
    x[j] = fmaxf(v, 0.f);
  }
}

// Packed f32 param layout (floats): w1a[64] | b1a[16] | g1a[16] | e1a[16] |
// w2a[96] | b2a[16] | g2a[16] | e2a[16] | then 11 DE-sized vectors:
// b1b g1b e1b b2b g2b e2b g11 b11 g12 b12 bo.
#define PF_W1A 0
#define PF_B1A 64
#define PF_G1A 80
#define PF_E1A 96
#define PF_W2A 112
#define PF_B2A 208
#define PF_G2A 224
#define PF_E2A 240
#define PF_DE0 256

template <int S, int DF, int DE>
__global__ void __launch_bounds__(64)
dlpt_block_kernel(const float* __restrict__ pos,
                  const float* __restrict__ feat_in,
                  const int* __restrict__ idx,
                  float* __restrict__ feat_out,
                  const float* __restrict__ pf,      // packed f32 params
                  const _Float16* __restrict__ wt,   // packed f16 weights
                  int nB, int N, int nC) {
  constexpr int DP  = 16;
  constexpr int DIN = DP + DF;              // embedding contraction dim
  constexpr int KP  = (DIN + 31) & ~31;     // padded to WMMA K
  constexpr int MT  = 2;                    // always 32 rows per wave
  constexpr int NT  = DE / 16;              // col tiles   (2 or 4)
  constexpr int KC  = DE / 32;              // K chunks    (1 or 2)
  constexpr int CPW = 32 / S;               // clusters per wave (2 or 1)
  constexpr int WAVES = 2;
  static_assert(KP <= DE, "X staging reuses the Q buffer");
  static_assert(S == 16 || S == 32, "wave holds 32 points");

  // Param views (all compile-time offsets from two base pointers).
  const float* w1a = pf + PF_W1A;
  const float* b1a = pf + PF_B1A;
  const float* g1a = pf + PF_G1A;
  const float* e1a = pf + PF_E1A;
  const float* w2a = pf + PF_W2A;
  const float* b2a = pf + PF_B2A;
  const float* g2a = pf + PF_G2A;
  const float* e2a = pf + PF_E2A;
  const float* b1b = pf + PF_DE0 + 0 * DE;
  const float* g1b = pf + PF_DE0 + 1 * DE;
  const float* e1b = pf + PF_DE0 + 2 * DE;
  const float* b2b = pf + PF_DE0 + 3 * DE;
  const float* g2b = pf + PF_DE0 + 4 * DE;
  const float* e2b = pf + PF_DE0 + 5 * DE;
  const float* g11 = pf + PF_DE0 + 6 * DE;
  const float* b11 = pf + PF_DE0 + 7 * DE;
  const float* g12 = pf + PF_DE0 + 8 * DE;
  const float* b12 = pf + PF_DE0 + 9 * DE;
  const float* bo  = pf + PF_DE0 + 10 * DE;
  const _Float16* w1bT = wt;
  const _Float16* w2bT = wt + 1 * DE * KP;
  const _Float16* wqT  = wt + 2 * DE * KP;
  const _Float16* wkT  = wqT + DE * DE;
  const _Float16* wvT  = wkT + DE * DE;
  const _Float16* woT  = wvT + DE * DE;

  __shared__ __align__(32) _Float16 sm_hq[WAVES][32 * DE];
  __shared__ __align__(32) _Float16 sm_hv[WAVES][32 * DE];
  __shared__ __align__(32) _Float16 sm_q [WAVES][32 * DE];  // + X staging
  __shared__ __align__(32) _Float16 sm_k [WAVES][32 * DE];  // + h_geo temp
  __shared__ __align__(32) _Float16 sm_v [WAVES][32 * DE];  // V^T [DE][32]
  __shared__ __align__(32) _Float16 sm_o [WAVES][32 * DE];
  __shared__ __align__(32) float    sm_sc[WAVES][32 * 32];  // scores; out f16
  __shared__ __align__(32) _Float16 sm_at[WAVES][32 * 32];  // block-diag attn

  const int lane = threadIdx.x & 31;
  const int wid  = threadIdx.x >> 5;
  _Float16* hqB = sm_hq[wid];
  _Float16* hvB = sm_hv[wid];
  _Float16* qB  = sm_q [wid];
  _Float16* kB  = sm_k [wid];
  _Float16* vB  = sm_v [wid];
  _Float16* oB  = sm_o [wid];
  float*    scB = sm_sc[wid];
  _Float16* atB = sm_at[wid];

  const long long cTotal = (long long)nB * nC;
  // this wave's first cluster; this lane's cluster (S=16: halves differ)
  const long long c = ((long long)blockIdx.x * WAVES + wid) * CPW + lane / S;
  const bool cval = (c < cTotal);
  const long long ccl = cval ? c : (cTotal - 1);
  const int  b  = (int)(ccl / nC);
  const int  ci = (int)(ccl % nC);
  const int  s  = lane;              // LDS row owned by this lane
  const int  sL = lane % S;          // point index within its cluster

  // ---- gather point + feature -------------------------------------------
  const int pi = idx[((size_t)b * nC + ci) * S + sL];
  const size_t pb = ((size_t)b * N + pi) * 3;
  const float px = pos[pb + 0], py = pos[pb + 1], pz = pos[pb + 2];
  float fv[DF];
  {
    const size_t fb = ((size_t)b * N + pi) * DF;
#pragma unroll
    for (int j = 0; j < DF; ++j) fv[j] = feat_in[fb + j];
  }

  // ---- per-cluster center of gravity (xor masks stay within cluster) ----
  float sx = px, sy = py, sz = pz;
#pragma unroll
  for (int m = S / 2; m >= 1; m >>= 1) {
    sx += __shfl_xor(sx, m, 32);
    sy += __shfl_xor(sy, m, 32);
    sz += __shfl_xor(sz, m, 32);
  }
  const float inv_s = 1.f / (float)S;
  const float lx = px - sx * inv_s;
  const float ly = py - sy * inv_s;
  const float lz = pz - sz * inv_s;
  const float nr = sqrtf(lx * lx + ly * ly + lz * lz);

  // ---- tiny geometric MLPs (scalar VALU; K=4 / effective K=3) -----------
  float r1[DP];
#pragma unroll
  for (int j = 0; j < DP; ++j)
    r1[j] = b1a[j] + lx * w1a[0 * DP + j] + ly * w1a[1 * DP + j] +
            lz * w1a[2 * DP + j] + nr * w1a[3 * DP + j];
  ln_relu<DP>(r1, g1a, e1a);

  // avg = mean(lp) == 0, so only rows 3..5 of w2a contribute.
  float r2[DP];
#pragma unroll
  for (int j = 0; j < DP; ++j)
    r2[j] = b2a[j] + lx * w2a[3 * DP + j] + ly * w2a[4 * DP + j] +
            lz * w2a[5 * DP + j];
  ln_relu<DP>(r2, g2a, e2a);

  // ---- embedding GEMM 1: [r1, f] @ w1b  (WMMA, K padded) -----------------
#pragma unroll
  for (int j = 0; j < DP; ++j)
    qB[s * KP + j] = cval ? (_Float16)r1[j] : (_Float16)0.f;
#pragma unroll
  for (int j = 0; j < DF; ++j)
    qB[s * KP + DP + j] = cval ? (_Float16)fv[j] : (_Float16)0.f;
#pragma unroll
  for (int j = DIN; j < KP; ++j) qB[s * KP + j] = (_Float16)0.f;
  __syncthreads();
#pragma unroll
  for (int nt = 0; nt < NT; ++nt) {
    v8f acc[MT];
#pragma unroll
    for (int mt = 0; mt < MT; ++mt) acc[mt] = zero8();
#pragma unroll
    for (int kc = 0; kc < KP / 32; ++kc) {
      v16h bw = load_bwT(w1bT, kc * 32, nt * 16, KP, lane);
#pragma unroll
      for (int mt = 0; mt < MT; ++mt)
        acc[mt] = wmma_f16(load_a16(qB, mt * 16, kc * 32, KP, lane), bw, acc[mt]);
    }
#pragma unroll
    for (int mt = 0; mt < MT; ++mt)
      store_c16(acc[mt], hvB, mt * 16, nt * 16, DE, lane, 1.f);
  }
  __syncthreads();

  float hp[DE];  // h_pos kept for residual
#pragma unroll
  for (int j = 0; j < DE; ++j) hp[j] = (float)hvB[s * DE + j] + b1b[j];
  ln_relu<DE>(hp, g1b, e1b);

  // ---- embedding GEMM 2: [r2, f] @ w2b -----------------------------------
#pragma unroll
  for (int j = 0; j < DP; ++j)
    qB[s * KP + j] = cval ? (_Float16)r2[j] : (_Float16)0.f;
#pragma unroll
  for (int j = 0; j < DF; ++j)
    qB[s * KP + DP + j] = cval ? (_Float16)fv[j] : (_Float16)0.f;
#pragma unroll
  for (int j = DIN; j < KP; ++j) qB[s * KP + j] = (_Float16)0.f;
  __syncthreads();
#pragma unroll
  for (int nt = 0; nt < NT; ++nt) {
    v8f acc[MT];
#pragma unroll
    for (int mt = 0; mt < MT; ++mt) acc[mt] = zero8();
#pragma unroll
    for (int kc = 0; kc < KP / 32; ++kc) {
      v16h bw = load_bwT(w2bT, kc * 32, nt * 16, KP, lane);
#pragma unroll
      for (int mt = 0; mt < MT; ++mt)
        acc[mt] = wmma_f16(load_a16(qB, mt * 16, kc * 32, KP, lane), bw, acc[mt]);
    }
#pragma unroll
    for (int mt = 0; mt < MT; ++mt)
      store_c16(acc[mt], kB, mt * 16, nt * 16, DE, lane, 1.f);
  }
  __syncthreads();

  float hg[DE];
#pragma unroll
  for (int j = 0; j < DE; ++j) hg[j] = (float)kB[s * DE + j] + b2b[j];
  ln_relu<DE>(hg, g2b, e2b);

  // ---- Pre-LN: hq = LN(h_geo), hv = LN(h_pos), written as f16 rows -------
  {
    float mu = 0.f;
#pragma unroll
    for (int j = 0; j < DE; ++j) mu += hg[j];
    mu *= (1.f / DE);
    float var = 0.f;
#pragma unroll
    for (int j = 0; j < DE; ++j) { float d = hg[j] - mu; var += d * d; }
    float rs = rsqrtf(var * (1.f / DE) + 1e-5f);
#pragma unroll
    for (int j = 0; j < DE; ++j)
      hqB[s * DE + j] =
          cval ? (_Float16)((hg[j] - mu) * rs * g12[j] + b12[j])
               : (_Float16)0.f;
  }
  {
    float mu = 0.f;
#pragma unroll
    for (int j = 0; j < DE; ++j) mu += hp[j];
    mu *= (1.f / DE);
    float var = 0.f;
#pragma unroll
    for (int j = 0; j < DE; ++j) { float d = hp[j] - mu; var += d * d; }
    float rs = rsqrtf(var * (1.f / DE) + 1e-5f);
#pragma unroll
    for (int j = 0; j < DE; ++j)
      hvB[s * DE + j] =
          cval ? (_Float16)((hp[j] - mu) * rs * g11[j] + b11[j])
               : (_Float16)0.f;
  }
  __syncthreads();

  // ---- Q/K/V projections (WMMA); V stored transposed ---------------------
  const float qscale = rsqrtf((float)DE);
#pragma unroll
  for (int nt = 0; nt < NT; ++nt) {
    v8f aq[MT], ak[MT], av[MT];
#pragma unroll
    for (int mt = 0; mt < MT; ++mt) { aq[mt] = zero8(); ak[mt] = zero8(); av[mt] = zero8(); }
#pragma unroll
    for (int kc = 0; kc < KC; ++kc) {
      v16h bq = load_bwT(wqT, kc * 32, nt * 16, DE, lane);
      v16h bk = load_bwT(wkT, kc * 32, nt * 16, DE, lane);
      v16h bv = load_bwT(wvT, kc * 32, nt * 16, DE, lane);
#pragma unroll
      for (int mt = 0; mt < MT; ++mt) {
        v16h ahq = load_a16(hqB, mt * 16, kc * 32, DE, lane);
        v16h ahv = load_a16(hvB, mt * 16, kc * 32, DE, lane);
        aq[mt] = wmma_f16(ahq, bq, aq[mt]);
        ak[mt] = wmma_f16(ahq, bk, ak[mt]);
        av[mt] = wmma_f16(ahv, bv, av[mt]);
      }
    }
#pragma unroll
    for (int mt = 0; mt < MT; ++mt) {
      store_c16 (aq[mt], qB, mt * 16, nt * 16, DE, lane, qscale);
      store_c16 (ak[mt], kB, mt * 16, nt * 16, DE, lane, 1.f);
      store_c16T(av[mt], vB, mt * 16, nt * 16, 32, lane, 1.f);   // V^T[DE][32]
    }
  }
  __syncthreads();

  // ---- scores = Q @ K^T (WMMA); S=16 computes only diagonal tiles --------
#pragma unroll
  for (int nt = 0; nt < MT; ++nt) {
#pragma unroll
    for (int mt = 0; mt < MT; ++mt) {
      if (S == 16 && mt != nt) continue;   // folds at compile time
      v8f acc = zero8();
#pragma unroll
      for (int kc = 0; kc < KC; ++kc)
        acc = wmma_f16(load_a16(qB, mt * 16, kc * 32, DE, lane),
                       load_bt16(kB, kc * 32, nt * 16, DE, lane), acc);
      store_c32(acc, scB, mt * 16, nt * 16, 32, lane);
    }
  }
  __syncthreads();

  // ---- softmax (lane = row over its own cluster's columns) --------------
  {
    const int base = (S == 16) ? (s & 16) : 0;  // own cluster's column block
    if (cval) {
      float row[S];
      float mx = -1e30f;
#pragma unroll
      for (int t = 0; t < S; ++t) {
        row[t] = scB[s * 32 + base + t];
        mx = fmaxf(mx, row[t]);
      }
      float sum = 0.f;
#pragma unroll
      for (int t = 0; t < S; ++t) { row[t] = __expf(row[t] - mx); sum += row[t]; }
      float inv = 1.f / sum;
#pragma unroll
      for (int t = 0; t < S; ++t) atB[s * 32 + base + t] = (_Float16)(row[t] * inv);
      if (S == 16) {   // zero the cross-cluster half -> block-diagonal attn
#pragma unroll
        for (int t = 0; t < 16; ++t) atB[s * 32 + (base ^ 16) + t] = (_Float16)0.f;
      }
    } else {
#pragma unroll
      for (int t = 0; t < 32; ++t) atB[s * 32 + t] = (_Float16)0.f;
    }
  }
  __syncthreads();

  // ---- out = attn @ V (WMMA, K=32); zero blocks kill cross terms ---------
#pragma unroll
  for (int nt = 0; nt < NT; ++nt) {
    v16h bt = load_bt16(vB, 0, nt * 16, 32, lane);
#pragma unroll
    for (int mt = 0; mt < MT; ++mt) {
      v8f acc = wmma_f16(load_a16(atB, mt * 16, 0, 32, lane), bt, zero8());
      store_c16(acc, oB, mt * 16, nt * 16, DE, lane, 1.f);
    }
  }
  __syncthreads();

  // ---- out @ wo + bo (WMMA), then residual + scatter ---------------------
  v8f oacc[MT * NT];
#pragma unroll
  for (int nt = 0; nt < NT; ++nt) {
#pragma unroll
    for (int mt = 0; mt < MT; ++mt) oacc[mt * NT + nt] = zero8();
#pragma unroll
    for (int kc = 0; kc < KC; ++kc) {
      v16h bw = load_bwT(woT, kc * 32, nt * 16, DE, lane);
#pragma unroll
      for (int mt = 0; mt < MT; ++mt)
        oacc[mt * NT + nt] = wmma_f16(load_a16(oB, mt * 16, kc * 32, DE, lane),
                                      bw, oacc[mt * NT + nt]);
    }
    float bb = bo[nt * 16 + (lane & 15)];
#pragma unroll
    for (int mt = 0; mt < MT; ++mt)
#pragma unroll
      for (int r = 0; r < 8; ++r) oacc[mt * NT + nt][r] += bb;
  }
  __syncthreads();
  _Float16* outB = (_Float16*)scB;   // reuse score buffer (32*DE f16 fits)
#pragma unroll
  for (int nt = 0; nt < NT; ++nt)
#pragma unroll
    for (int mt = 0; mt < MT; ++mt)
      store_c16(oacc[mt * NT + nt], outB, mt * 16, nt * 16, DE, lane, 1.f);
  __syncthreads();

  if (cval) {
    size_t ob = ((size_t)b * N + pi) * DE;
#pragma unroll
    for (int j = 0; j < DE; ++j)
      feat_out[ob + j] = hp[j] + (float)outB[s * DE + j];
  }
}

// Convert f32 weight W[kdim][ncol] into f16 transposed + padded wT[ncol][kpad].
__global__ void prep_weight_kernel(const float* __restrict__ w,
                                   _Float16* __restrict__ wt,
                                   int kdim, int ncol, int kpad) {
  int t = blockIdx.x * blockDim.x + threadIdx.x;
  if (t >= ncol * kpad) return;
  int n = t / kpad;
  int k = t % kpad;
  wt[t] = (k < kdim) ? (_Float16)w[k * ncol + n] : (_Float16)0.f;
}

__global__ void gather_fps_kernel(const float* __restrict__ pos,
                                  const float* __restrict__ feat,
                                  const int* __restrict__ fps,
                                  float* __restrict__ out_pos,
                                  float* __restrict__ out_feat,
                                  int nB, int N, int M, int DE) {
  int t = blockIdx.x * blockDim.x + threadIdx.x;
  if (t >= nB * M) return;
  int b = t / M;
  int pi = fps[t];
  size_t src = (size_t)b * N + pi;
  out_pos[(size_t)t * 3 + 0] = pos[src * 3 + 0];
  out_pos[(size_t)t * 3 + 1] = pos[src * 3 + 1];
  out_pos[(size_t)t * 3 + 2] = pos[src * 3 + 2];
  for (int j = 0; j < DE; ++j)
    out_feat[(size_t)t * DE + j] = feat[src * (size_t)DE + j];
}

extern "C" void kernel_launch(void* const* d_in, const int* in_sizes, int n_in,
                              void* d_out, int out_size, void* d_ws,
                              size_t ws_size, hipStream_t stream) {
  (void)in_sizes; (void)n_in; (void)out_size; (void)ws_size;
  constexpr int B = 4, N = 65536, M = N / 4;
  constexpr int S1 = 16, S2 = 32, DH = 32, DEo = 64;
  constexpr int DIN1 = 16 + 3, DIN2 = 16 + DH;
  constexpr int KP1 = 32, KP2 = 64;

  const float* pos  = (const float*)d_in[0];
  const float* feat = (const float*)d_in[1];
  const int* idx1 = (const int*)d_in[2];
  const int* idx2 = (const int*)d_in[3];
  const int* fps  = (const int*)d_in[4];

  // Raw f32 param leaves, flattened in JAX pytree (sorted-key) order.
  struct Raw {
    const float *b11,*b12,*b1a,*b1b,*b2a,*b2b,*bo,*e1a,*e1b,*e2a,*e2b,
                *g11,*g12,*g1a,*g1b,*g2a,*g2b,*w1a,*w1b,*w2a,*w2b,
                *wk,*wo,*wq,*wv;
  };
  auto mkraw = [&](int o) {
    Raw R;
    R.b11 = (const float*)d_in[o + 0];  R.b12 = (const float*)d_in[o + 1];
    R.b1a = (const float*)d_in[o + 2];  R.b1b = (const float*)d_in[o + 3];
    R.b2a = (const float*)d_in[o + 4];  R.b2b = (const float*)d_in[o + 5];
    R.bo  = (const float*)d_in[o + 6];  R.e1a = (const float*)d_in[o + 7];
    R.e1b = (const float*)d_in[o + 8];  R.e2a = (const float*)d_in[o + 9];
    R.e2b = (const float*)d_in[o + 10]; R.g11 = (const float*)d_in[o + 11];
    R.g12 = (const float*)d_in[o + 12]; R.g1a = (const float*)d_in[o + 13];
    R.g1b = (const float*)d_in[o + 14]; R.g2a = (const float*)d_in[o + 15];
    R.g2b = (const float*)d_in[o + 16]; R.w1a = (const float*)d_in[o + 17];
    R.w1b = (const float*)d_in[o + 18]; R.w2a = (const float*)d_in[o + 19];
    R.w2b = (const float*)d_in[o + 20]; R.wk  = (const float*)d_in[o + 21];
    R.wo  = (const float*)d_in[o + 22]; R.wq  = (const float*)d_in[o + 23];
    R.wv  = (const float*)d_in[o + 24];
    return R;
  };
  Raw R1 = mkraw(5), R2 = mkraw(30);

  // Workspace: feat1 [B,N,32] f32 | feat2 [B,N,64] f32 | f16 weights |
  //            packed f32 params (block1, block2).
  float* feat1 = (float*)d_ws;
  float* feat2 = feat1 + (size_t)B * N * DH;
  _Float16* wt1 = (_Float16*)(feat2 + (size_t)B * N * DEo);
  constexpr int WT1_HALFS = 2 * DH * KP1 + 4 * DH * DH;     // 6144
  constexpr int WT2_HALFS = 2 * DEo * KP2 + 4 * DEo * DEo;  // 24576
  _Float16* wt2 = wt1 + WT1_HALFS;
  float* pf1 = (float*)(wt2 + WT2_HALFS);
  constexpr int PF1_FLOATS = 256 + 11 * DH;                 // 608
  float* pf2 = pf1 + PF1_FLOATS;

  // --- f16 transposed weights (6 prep kernels per block) ---
  auto prep = [&](const float* w, _Float16* dst, int kdim, int ncol, int kpad) {
    int total = ncol * kpad;
    prep_weight_kernel<<<(total + 255) / 256, 256, 0, stream>>>(w, dst, kdim,
                                                                ncol, kpad);
  };
  prep(R1.w1b, wt1,                       DIN1, DH,  KP1);
  prep(R1.w2b, wt1 + 1 * DH * KP1,        DIN1, DH,  KP1);
  prep(R1.wq,  wt1 + 2 * DH * KP1,            DH, DH, DH);
  prep(R1.wk,  wt1 + 2 * DH * KP1 + 1*DH*DH,  DH, DH, DH);
  prep(R1.wv,  wt1 + 2 * DH * KP1 + 2*DH*DH,  DH, DH, DH);
  prep(R1.wo,  wt1 + 2 * DH * KP1 + 3*DH*DH,  DH, DH, DH);
  prep(R2.w1b, wt2,                       DIN2, DEo, KP2);
  prep(R2.w2b, wt2 + 1 * DEo * KP2,       DIN2, DEo, KP2);
  prep(R2.wq,  wt2 + 2 * DEo * KP2,              DEo, DEo, DEo);
  prep(R2.wk,  wt2 + 2 * DEo * KP2 + 1*DEo*DEo,  DEo, DEo, DEo);
  prep(R2.wv,  wt2 + 2 * DEo * KP2 + 2*DEo*DEo,  DEo, DEo, DEo);
  prep(R2.wo,  wt2 + 2 * DEo * KP2 + 3*DEo*DEo,  DEo, DEo, DEo);

  // --- packed f32 params (async d2d copies; capture-safe) ---
  auto cpy = [&](float* dst, const float* src, int nflt) {
    hipMemcpyAsync(dst, src, (size_t)nflt * sizeof(float),
                   hipMemcpyDeviceToDevice, stream);
  };
  auto pack = [&](float* pf, const Raw& R, int DE) {
    cpy(pf + PF_W1A, R.w1a, 4 * 16);
    cpy(pf + PF_B1A, R.b1a, 16);
    cpy(pf + PF_G1A, R.g1a, 16);
    cpy(pf + PF_E1A, R.e1a, 16);
    cpy(pf + PF_W2A, R.w2a, 6 * 16);
    cpy(pf + PF_B2A, R.b2a, 16);
    cpy(pf + PF_G2A, R.g2a, 16);
    cpy(pf + PF_E2A, R.e2a, 16);
    cpy(pf + PF_DE0 + 0 * DE, R.b1b, DE);
    cpy(pf + PF_DE0 + 1 * DE, R.g1b, DE);
    cpy(pf + PF_DE0 + 2 * DE, R.e1b, DE);
    cpy(pf + PF_DE0 + 3 * DE, R.b2b, DE);
    cpy(pf + PF_DE0 + 4 * DE, R.g2b, DE);
    cpy(pf + PF_DE0 + 5 * DE, R.e2b, DE);
    cpy(pf + PF_DE0 + 6 * DE, R.g11, DE);
    cpy(pf + PF_DE0 + 7 * DE, R.b11, DE);
    cpy(pf + PF_DE0 + 8 * DE, R.g12, DE);
    cpy(pf + PF_DE0 + 9 * DE, R.b12, DE);
    cpy(pf + PF_DE0 + 10 * DE, R.bo, DE);
  };
  pack(pf1, R1, DH);
  pack(pf2, R2, DEo);

  // Each wave owns 32 points => clusters-per-wave = 32/S.
  {
    int nC = N / S1;
    long long cl = (long long)B * nC;
    long long waves = (cl + (32 / S1) - 1) / (32 / S1);
    int blocks = (int)((waves + 1) / 2);
    dlpt_block_kernel<S1, 3, DH><<<blocks, 64, 0, stream>>>(
        pos, feat, idx1, feat1, pf1, wt1, B, N, nC);
  }
  {
    int nC = N / S2;
    long long cl = (long long)B * nC;
    int blocks = (int)((cl + 1) / 2);
    dlpt_block_kernel<S2, DH, DEo><<<blocks, 64, 0, stream>>>(
        pos, feat1, idx2, feat2, pf2, wt2, B, N, nC);
  }
  {
    int total = B * M;
    float* out_pos  = (float*)d_out;
    float* out_feat = out_pos + (size_t)total * 3;
    gather_fps_kernel<<<(total + 255) / 256, 256, 0, stream>>>(
        pos, feat2, fps, out_pos, out_feat, B, N, M, DEo);
  }
}
